// Seq2Seq_75316546502715
// MI455X (gfx1250) — matrix operand order repaired and checked by reading, not compiled, not run on hardware
//
#include <hip/hip_runtime.h>

// ---------------------------------------------------------------------------
// Seq2Seq LSTM (B=2048, T=512, H=64, L=2, horizon=48) for gfx1250 (MI455X).
//
// Roofline: ~150 GFLOP total, 134 MB mandatory traffic -> compute/latency-
// bound serial recurrence. Strategy: weights resident in LDS as pre-swizzled
// bf16 WMMA B-fragments (128 KB), h-state in LDS bf16, c-state in VGPRs, all
// gate GEMMs on v_wmma_f32_16x16x32_bf16 (fp32 accumulate). One block = 16
// batch rows, 8 waves; wave w owns gate N-tiles {2w,2w+1}.
// This rev: f32->bf16 conversions routed through compiler-native fptrunc
// (__builtin_convertvector / __bf16 cast) so the backend can select a native
// cvt op if gfx1250 has one; hardware v_tanh_f32 nonlinearities retained.
// ---------------------------------------------------------------------------

#define F_IN    32
#define F_OUT   8
#define H       64
#define GATES   256        // 4*H, PyTorch gate order i,f,g,o
#define L       2
#define T_STEPS 512
#define HORIZON 48
#define BATCH   2048
#define BT      16         // batch rows per block
#define NT      16         // gate N-tiles (256/16)
#define KT      2          // K-tiles per GEMM (64/32)
#define WBLK    (32 * 512) // ushorts per (layer,matrix) fragment block (32 KB)

typedef __attribute__((ext_vector_type(16))) __bf16 v16bf;
typedef __attribute__((ext_vector_type(2)))  __bf16 v2bf;
typedef __attribute__((ext_vector_type(2)))  float  v2f;
typedef __attribute__((ext_vector_type(8)))  float  v8f;

union Frag {
  v16bf        v;
  unsigned int u32[8];
};

// ---- f32 -> bf16: prefer dedicated builtin, else compiler-native fptrunc --
#if __has_builtin(__builtin_amdgcn_cvt_pk_bf16_f32)
__device__ __forceinline__ unsigned int pk2bf(float lo, float hi) {
  v2bf r = __builtin_amdgcn_cvt_pk_bf16_f32(lo, hi);
  unsigned int u;
  __builtin_memcpy(&u, &r, 4);
  return u;
}
__device__ __forceinline__ unsigned short f2bf(float f) {
  return (unsigned short)(pk2bf(f, f) & 0xFFFFu);
}
#else
__device__ __forceinline__ unsigned int pk2bf(float lo, float hi) {
  v2f s = {lo, hi};
  v2bf r = __builtin_convertvector(s, v2bf);   // fptrunc: native cvt if present
  unsigned int u;
  __builtin_memcpy(&u, &r, 4);
  return u;
}
__device__ __forceinline__ unsigned short f2bf(float f) {
  __bf16 b = (__bf16)f;
  unsigned short u;
  __builtin_memcpy(&u, &b, 2);
  return u;
}
#endif

__device__ __forceinline__ float bf2f(unsigned short h) {
  return __uint_as_float(((unsigned int)h) << 16);
}

// ---- nonlinearities: prefer hardware V_TANH_F32 (gfx1250 trans op) --------
__device__ __forceinline__ float tanhx(float x) {
#if __has_builtin(__builtin_amdgcn_tanhf)
  return __builtin_amdgcn_tanhf(x);
#elif __has_builtin(__builtin_amdgcn_tanh_f32)
  return __builtin_amdgcn_tanh_f32(x);
#else
  float e = __expf(-2.0f * x);
  return (1.0f - e) / (1.0f + e);
#endif
}
__device__ __forceinline__ float sigf(float x) {
#if __has_builtin(__builtin_amdgcn_tanhf) || __has_builtin(__builtin_amdgcn_tanh_f32)
  return __builtin_fmaf(0.5f, tanhx(0.5f * x), 0.5f);  // sig(x)=0.5*tanh(x/2)+0.5
#else
  return 1.0f / (1.0f + __expf(-x));
#endif
}

// --- Pack global f32 weight matrix W[n][k] (row stride ks) into bf16 WMMA
//     B-fragments (B[k][n] = W[n][k]); fragment order [tn][tk], 1 KB each.
//     B layout (16x16x32): lanes 0-15 hold K {0..7,16..23}, lanes 16-31 K {8..15,24..31}.
__device__ void pack_wfrag(const float* __restrict__ W, unsigned short* dst,
                           int nt, int kt, int ks, int tid) {
  const int wv = tid >> 5, lane = tid & 31;
  const int nlo = lane & 15, half = lane >> 4;
  for (int fi = wv; fi < nt * kt; fi += 8) {
    const int tn = fi / kt, tk = fi % kt;
    const float* src = W + (tn * 16 + nlo) * ks + tk * 32 + half * 8;
    unsigned int* d = (unsigned int*)(dst + fi * 512 + lane * 16);
#pragma unroll
    for (int v = 0; v < 8; ++v) {
      float2 w2 = *(const float2*)(src + 2 * v + (v & 4) * 2);
      d[v] = pk2bf(w2.x, w2.y);
    }
  }
}

// --- A-fragment (16x32 bf16, M x K) from LDS [16][64] bf16 state buffer.
//     Lane holds row M=lane%16; K interleave {0..7,16..23 | 8..15,24..31}.
//     Contiguous dword groups -> compiler merges into ds_load_b128 pairs.
__device__ __forceinline__ Frag load_afrag(const unsigned short* src,
                                           int kblock, int nlo, int half) {
  Frag f;
  const unsigned int* r = (const unsigned int*)(src + nlo * H + kblock * 32);
#pragma unroll
  for (int v = 0; v < 4; ++v) {
    f.u32[v]     = r[v + 4 * half];
    f.u32[4 + v] = r[8 + v + 4 * half];
  }
  return f;
}

// --- B-fragment: 32 contiguous bytes per lane -> two ds_load_b128.
__device__ __forceinline__ Frag load_bfrag(const unsigned short* base,
                                           int fi, int lane) {
  Frag f;
  const uint4* p = (const uint4*)(base + fi * 512 + lane * 16);
  uint4 a = p[0], b = p[1];
  f.u32[0] = a.x; f.u32[1] = a.y; f.u32[2] = a.z; f.u32[3] = a.w;
  f.u32[4] = b.x; f.u32[5] = b.y; f.u32[6] = b.z; f.u32[7] = b.w;
  return f;
}

__device__ __forceinline__ v8f wmma_bf16(Frag a, Frag b, v8f c) {
  return __builtin_amdgcn_wmma_f32_16x16x32_bf16(false, a.v, false, b.v,
                                                 (short)0, c, false, false);
}

// --- Gate GEMM phase: g[16,256] = x[16,64] @ Wih^T + h[16,64] @ Whh^T + bias
//     Wave w computes N-tiles 2w, 2w+1 (8 WMMA); bias folded into C init.
__device__ __forceinline__ void gate_phase(const unsigned short* xsrc,
                                           const unsigned short* hsrc,
                                           const unsigned short* wih,
                                           const unsigned short* whh,
                                           float* gbuf, float bias0, float bias1,
                                           int wv, int lane, int nlo, int half) {
  Frag ax0 = load_afrag(xsrc, 0, nlo, half);
  Frag ax1 = load_afrag(xsrc, 1, nlo, half);
  Frag ah0 = load_afrag(hsrc, 0, nlo, half);
  Frag ah1 = load_afrag(hsrc, 1, nlo, half);
#pragma unroll
  for (int tt = 0; tt < 2; ++tt) {
    const int tn = 2 * wv + tt;
    const float bb = tt ? bias1 : bias0;
    v8f acc;
#pragma unroll
    for (int r = 0; r < 8; ++r) acc[r] = bb;
    acc = wmma_bf16(ax0, load_bfrag(wih, tn * KT + 0, lane), acc);
    acc = wmma_bf16(ax1, load_bfrag(wih, tn * KT + 1, lane), acc);
    acc = wmma_bf16(ah0, load_bfrag(whh, tn * KT + 0, lane), acc);
    acc = wmma_bf16(ah1, load_bfrag(whh, tn * KT + 1, lane), acc);
    float* gc = gbuf + tn * 16 + nlo;            // C/D layout: M=r+8*half
#pragma unroll
    for (int r = 0; r < 8; ++r) gc[(r + 8 * half) * GATES] = acc[r];
  }
}

// --- LSTM cell elementwise: thread owns (row em, 4 hidden cols at ec0);
//     c stays in VGPRs across the whole kernel.
__device__ __forceinline__ void cell_phase(const float* gbuf,
                                           unsigned short* hdst,
                                           float* c, int em, int ec0) {
  const float* gr = gbuf + em * GATES;
  float hv[4];
#pragma unroll
  for (int j = 0; j < 4; ++j) {
    float i_ = gr[ec0 + j];
    float f_ = gr[64 + ec0 + j];
    float g_ = gr[128 + ec0 + j];
    float o_ = gr[192 + ec0 + j];
    float cn = sigf(f_) * c[j] + sigf(i_) * tanhx(g_);
    c[j] = cn;
    hv[j] = sigf(o_) * tanhx(cn);
  }
  unsigned int* d = (unsigned int*)(hdst + em * H + ec0);
  d[0] = pk2bf(hv[0], hv[1]);
  d[1] = pk2bf(hv[2], hv[3]);
}

__global__ __launch_bounds__(256)
void seq2seq_lstm_kernel(const float* __restrict__ X,
                         const float* __restrict__ W_emb,
                         const float* __restrict__ b_emb,
                         const float* __restrict__ enc_Wih,
                         const float* __restrict__ enc_Whh,
                         const float* __restrict__ enc_bih,
                         const float* __restrict__ enc_bhh,
                         const float* __restrict__ dec_Wih,
                         const float* __restrict__ dec_Whh,
                         const float* __restrict__ dec_bih,
                         const float* __restrict__ dec_bhh,
                         const float* __restrict__ W_reg,
                         const float* __restrict__ b_reg,
                         float* __restrict__ out) {
  // ~156 KB static LDS (fits one workgroup within the 320 KB WGP pool)
  __shared__ __align__(16) unsigned short s_wfrag[4 * WBLK];      // 128 KB weights
  __shared__ __align__(16) unsigned short s_embfrag[4 * 512];     // 4 KB  W_emb
  __shared__ __align__(16) float          s_gbuf[BT * GATES];     // 16 KB gates f32
  __shared__ __align__(16) unsigned short s_hbuf[L * BT * H];     // 4 KB  h (bf16)
  __shared__ __align__(16) unsigned short s_xbuf[BT * H];         // 2 KB  layer-0 in
  __shared__ float s_wreg[F_OUT * H];                             // 2 KB
  __shared__ float s_breg[F_OUT];

  const int tid  = threadIdx.x;
  const int wv   = tid >> 5;
  const int lane = tid & 31;
  const int nlo  = lane & 15;
  const int half = lane >> 4;
  const int b0   = blockIdx.x * BT;
  const int em   = tid >> 4;           // elementwise row 0..15
  const int ec0  = (tid & 15) * 4;     // elementwise col base

  // ---- setup: pack encoder weights + embedding into WMMA fragments --------
#pragma unroll
  for (int l = 0; l < L; ++l) {
    pack_wfrag(enc_Wih + l * GATES * H, s_wfrag + (2 * l + 0) * WBLK, NT, KT, H, tid);
    pack_wfrag(enc_Whh + l * GATES * H, s_wfrag + (2 * l + 1) * WBLK, NT, KT, H, tid);
  }
  pack_wfrag(W_emb, s_embfrag, 4, 1, F_IN, tid);
  for (int i = tid; i < F_OUT * H; i += 256) s_wreg[i] = W_reg[i];
  if (tid < F_OUT) s_breg[tid] = b_reg[tid];
  for (int i = tid; i < L * BT * H; i += 256) s_hbuf[i] = 0;

  float biasr[L][2];
  {
    const int n0 = wv * 32 + nlo;
#pragma unroll
    for (int l = 0; l < L; ++l) {
      biasr[l][0] = enc_bih[l * GATES + n0]      + enc_bhh[l * GATES + n0];
      biasr[l][1] = enc_bih[l * GATES + n0 + 16] + enc_bhh[l * GATES + n0 + 16];
    }
  }
  const float bembr = (wv < 4) ? b_emb[wv * 16 + nlo] : 0.0f;
  float creg[L][4] = {};
  __syncthreads();

  // ---- encoder: 512 serial steps -----------------------------------------
  for (int t = 0; t < T_STEPS; ++t) {
    if (wv < 4) {  // embedding: emb[16,64] = X_t[16,32] @ W_emb^T + b_emb
      const float* xrow = X + ((size_t)(b0 + nlo) * T_STEPS + t) * F_IN;
      if (t + 8 < T_STEPS) __builtin_prefetch(xrow + 8 * F_IN, 0, 1);
      Frag ax;
#pragma unroll
      for (int v = 0; v < 4; ++v) {   // A K-interleave, f32 -> bf16 on the fly
        float2 a = *(const float2*)(xrow + 2 * v + 8 * half);
        float2 b = *(const float2*)(xrow + 16 + 2 * v + 8 * half);
        ax.u32[v]     = pk2bf(a.x, a.y);
        ax.u32[4 + v] = pk2bf(b.x, b.y);
      }
      v8f acc;
#pragma unroll
      for (int r = 0; r < 8; ++r) acc[r] = bembr;
      acc = wmma_bf16(ax, load_bfrag(s_embfrag, wv, lane), acc);
      unsigned short* xc = s_xbuf + wv * 16 + nlo;
#pragma unroll
      for (int r = 0; r < 8; ++r) xc[(r + 8 * half) * H] = f2bf(acc[r]);
    }
    __syncthreads();
    gate_phase(s_xbuf, s_hbuf, s_wfrag, s_wfrag + WBLK, s_gbuf,
               biasr[0][0], biasr[0][1], wv, lane, nlo, half);
    __syncthreads();
    cell_phase(s_gbuf, s_hbuf, creg[0], em, ec0);
    __syncthreads();
    gate_phase(s_hbuf, s_hbuf + BT * H, s_wfrag + 2 * WBLK, s_wfrag + 3 * WBLK,
               s_gbuf, biasr[1][0], biasr[1][1], wv, lane, nlo, half);
    __syncthreads();
    cell_phase(s_gbuf, s_hbuf + BT * H, creg[1], em, ec0);
    __syncthreads();
  }

  // ---- swap in decoder weights (reuse same LDS region) --------------------
#pragma unroll
  for (int l = 0; l < L; ++l) {
    pack_wfrag(dec_Wih + l * GATES * H, s_wfrag + (2 * l + 0) * WBLK, NT, KT, H, tid);
    pack_wfrag(dec_Whh + l * GATES * H, s_wfrag + (2 * l + 1) * WBLK, NT, KT, H, tid);
    const int n0 = wv * 32 + nlo;
    biasr[l][0] = dec_bih[l * GATES + n0]      + dec_bhh[l * GATES + n0];
    biasr[l][1] = dec_bih[l * GATES + n0 + 16] + dec_bhh[l * GATES + n0 + 16];
#pragma unroll
    for (int j = 0; j < 4; ++j) creg[l][j] = 0.0f;  // c_dec starts at zero
  }
  __syncthreads();

  // ---- decoder: 48 steps, x_t = previous top-layer h ----------------------
  for (int s = 0; s < HORIZON; ++s) {
    {
      uint2*       d  = (uint2*)s_xbuf;
      const uint2* sp = (const uint2*)(s_hbuf + BT * H);
      d[tid] = sp[tid];                       // copy h[top] -> layer-0 input
    }
    __syncthreads();
    gate_phase(s_xbuf, s_hbuf, s_wfrag, s_wfrag + WBLK, s_gbuf,
               biasr[0][0], biasr[0][1], wv, lane, nlo, half);
    __syncthreads();
    cell_phase(s_gbuf, s_hbuf, creg[0], em, ec0);
    __syncthreads();
    gate_phase(s_hbuf, s_hbuf + BT * H, s_wfrag + 2 * WBLK, s_wfrag + 3 * WBLK,
               s_gbuf, biasr[1][0], biasr[1][1], wv, lane, nlo, half);
    __syncthreads();
    cell_phase(s_gbuf, s_hbuf + BT * H, creg[1], em, ec0);
    __syncthreads();
    if (tid < BT * F_OUT) {   // y[16,8] = h_top @ W_reg^T + b_reg
      const int m = tid >> 3, fo = tid & 7;
      const unsigned short* hr = s_hbuf + BT * H + m * H;
      const float* wr = s_wreg + fo * H;
      float acc = s_breg[fo];
#pragma unroll 8
      for (int k = 0; k < H; ++k) acc += bf2f(hr[k]) * wr[k];
      out[((size_t)(b0 + m) * HORIZON + s) * F_OUT + fo] = acc;
    }
    __syncthreads();
  }
}

extern "C" void kernel_launch(void* const* d_in, const int* in_sizes, int n_in,
                              void* d_out, int out_size, void* d_ws, size_t ws_size,
                              hipStream_t stream) {
  (void)in_sizes; (void)n_in; (void)out_size; (void)d_ws; (void)ws_size;
  const float* X       = (const float*)d_in[0];
  /* d_in[1] = X_mask: all-ones, unused by the reference computation */
  const float* W_emb   = (const float*)d_in[2];
  const float* b_emb   = (const float*)d_in[3];
  const float* enc_Wih = (const float*)d_in[4];
  const float* enc_Whh = (const float*)d_in[5];
  const float* enc_bih = (const float*)d_in[6];
  const float* enc_bhh = (const float*)d_in[7];
  const float* dec_Wih = (const float*)d_in[8];
  const float* dec_Whh = (const float*)d_in[9];
  const float* dec_bih = (const float*)d_in[10];
  const float* dec_bhh = (const float*)d_in[11];
  const float* W_reg   = (const float*)d_in[12];
  const float* b_reg   = (const float*)d_in[13];
  float* out = (float*)d_out;

  seq2seq_lstm_kernel<<<dim3(BATCH / BT), dim3(256), 0, stream>>>(
      X, W_emb, b_emb, enc_Wih, enc_Whh, enc_bih, enc_bhh,
      dec_Wih, dec_Whh, dec_bih, dec_bhh, W_reg, b_reg, out);
}
